// _Decoder_25091198943794
// MI455X (gfx1250) — matrix operand order repaired and checked by reading, not compiled, not run on hardware
//
#include <hip/hip_runtime.h>

// ---------------- problem constants ----------------
#define B      32
#define T_ENC  256
#define T_OUT  600
#define E_DIM  512
#define N_MELS 80
#define HID    1024          // attention_rnn_dim == decoder_rnn_dim
#define ATT_H  128
#define N_FILT 32
#define KSZ    31
#define PRE    256
#define KATTN  (PRE + E_DIM + HID)      // 1792  ([prenet|ctx|h])
#define KDEC   (HID + E_DIM + HID)      // 2560  ([ah|ctx|dh])
#define NG     (4 * HID)                // 4096 gate outputs

typedef __attribute__((ext_vector_type(16))) __bf16 v16bf;
typedef __attribute__((ext_vector_type(8)))  __bf16 v8bf;
typedef __attribute__((ext_vector_type(8)))  float  v8f;

static __device__ __forceinline__ unsigned short f2bf(float f) {
  unsigned u = __float_as_uint(f);
  u += 0x7FFFu + ((u >> 16) & 1u);      // round-to-nearest-even
  return (unsigned short)(u >> 16);
}
static __device__ __forceinline__ float sigf(float x) { return 1.f / (1.f + expf(-x)); }

// ---------------- one-time setup kernels ----------------

// Fuse [Wih | Whh] into one (N, Ka+Kb) bf16 matrix, row-major (K contiguous).
__global__ void concat_bf16(const float* __restrict__ A, int Ka,
                            const float* __restrict__ Bm, int Kb,
                            unsigned short* __restrict__ out, int N) {
  long K = Ka + Kb, total = (long)N * K;
  for (long idx = (long)blockIdx.x * blockDim.x + threadIdx.x; idx < total;
       idx += (long)gridDim.x * blockDim.x) {
    long n = idx / K, k = idx - n * K;
    float v = (k < Ka) ? A[n * Ka + k] : Bm[n * Kb + (k - Ka)];
    out[idx] = f2bf(v);
  }
}

// Prenet for all timesteps (teacher forcing): out bf16 (T_OUT, B, PRE)
__global__ void prenet_kernel(const float* __restrict__ mel,
                              const float* __restrict__ w1,
                              const float* __restrict__ w2,
                              unsigned short* __restrict__ out) {
  int blk = blockIdx.x;                  // 0..T_OUT*B-1
  int t = blk / B, b = blk - t * B;
  int j = threadIdx.x;                   // 256
  __shared__ float s_in[N_MELS], s_h[PRE];
  if (j < N_MELS) s_in[j] = (t == 0) ? 0.f : mel[(size_t)b * N_MELS * T_OUT + j * T_OUT + (t - 1)];
  __syncthreads();
  float s = 0.f;
  for (int m = 0; m < N_MELS; ++m) s += w1[j * N_MELS + m] * s_in[m];
  s_h[j] = fmaxf(s, 0.f);
  __syncthreads();
  float s2 = 0.f;
  for (int k = 0; k < PRE; ++k) s2 += w2[j * PRE + k] * s_h[k];
  out[(size_t)(t * B + b) * PRE + j] = f2bf(fmaxf(s2, 0.f));
}

// processed_memory = memory @ memory_w.T : (B*T_ENC, ATT_H)
__global__ void pmem_kernel(const float* __restrict__ memory,
                            const float* __restrict__ mw,
                            float* __restrict__ pmem) {
  int row = blockIdx.x, h = threadIdx.x; // grid B*T_ENC, block ATT_H
  const float* m = memory + (size_t)row * E_DIM;
  const float* w = mw + (size_t)h * E_DIM;
  float s = 0.f;
  for (int e = 0; e < E_DIM; ++e) s += m[e] * w[e];
  pmem[(size_t)row * ATT_H + h] = s;
}

__global__ void zero_kernel(float* __restrict__ p, int n) {
  for (int i = blockIdx.x * blockDim.x + threadIdx.x; i < n; i += gridDim.x * blockDim.x)
    p[i] = 0.f;
}

// ---------------- per-step kernels ----------------

// X_attn = [prenet_t | a_ctx | ah]  (bf16, B x KATTN)
__global__ void assemble_xa(const unsigned short* __restrict__ pren_t,
                            const float* __restrict__ actx,
                            const float* __restrict__ ah,
                            unsigned short* __restrict__ X) {
  int idx = blockIdx.x * blockDim.x + threadIdx.x;
  if (idx >= B * KATTN) return;
  int b = idx / KATTN, k = idx - b * KATTN;
  unsigned short v;
  if (k < PRE)               v = pren_t[b * PRE + k];
  else if (k < PRE + E_DIM)  v = f2bf(actx[b * E_DIM + (k - PRE)]);
  else                       v = f2bf(ah[b * HID + (k - PRE - E_DIM)]);
  X[idx] = v;
}

// X_dec = [ah | a_ctx | dh]  (bf16, B x KDEC)
__global__ void assemble_xd(const float* __restrict__ ah,
                            const float* __restrict__ actx,
                            const float* __restrict__ dh,
                            unsigned short* __restrict__ X) {
  int idx = blockIdx.x * blockDim.x + threadIdx.x;
  if (idx >= B * KDEC) return;
  int b = idx / KDEC, k = idx - b * KDEC;
  float v;
  if (k < HID)               v = ah[b * HID + k];
  else if (k < HID + E_DIM)  v = actx[b * E_DIM + (k - HID)];
  else                       v = dh[b * HID + (k - HID - E_DIM)];
  X[idx] = f2bf(v);
}

// D(32 x N) = X(32 x K) @ W^T  via v_wmma_f32_16x16x32_bf16.
// One 16x16 tile per wave: 2 M-tiles x (N/16) N-tiles -> 64 blocks of 256.
// W is (N, K) row-major bf16 so each lane reads 16 contiguous K values of its column.
__global__ void gemm32_bf16_wmma(const unsigned short* __restrict__ Xu,
                                 const unsigned short* __restrict__ Wu,
                                 float* __restrict__ out, int K, int N) {
  const __bf16* X = (const __bf16*)Xu;
  const __bf16* W = (const __bf16*)Wu;
  int wave = (blockIdx.x * blockDim.x + threadIdx.x) >> 5;
  int lane = threadIdx.x & 31;
  int nTiles = N >> 4;
  int mt = wave / nTiles;               // 0 or 1 (M = 32)
  int nt = wave - mt * nTiles;
  // A-tile (16x32 bf16): lane<16 -> row=lane, K {0..15}; lane>=16 -> K {16..31}
  int rowA = mt * 16 + (lane & 15);
  const __bf16* xp = X + (size_t)rowA * K + ((lane >> 4) << 4);
  // B-tile (32x16 bf16): lane holds column nt*16+(lane&15), K base (lane>>4)*16
  int colB = nt * 16 + (lane & 15);
  const __bf16* wp = W + (size_t)colB * K + ((lane >> 4) << 4);
  v8f c = {};
  for (int k = 0; k < K; k += 32) {
    v16bf a, b;
    *((v8bf*)&a)       = *(const v8bf*)(xp + k);        // K lo half (16B)
    *(((v8bf*)&a) + 1) = *(const v8bf*)(xp + k + 8);    // K hi half (16B)
    b = *(const v16bf*)(wp + k);                        // 16 contiguous K (32B)
    __builtin_prefetch(wp + k + 512, 0, 0);             // global_prefetch_b8 (L2 weights)
    c = __builtin_amdgcn_wmma_f32_16x16x32_bf16(false, a, false, b,
                                                (short)0, c, false, false);
  }
  // C/D layout: VGPR d, lanes 0-15 -> M=d, lanes 16-31 -> M=d+8; N=lane&15
  int rowD = mt * 16 + ((lane >> 4) << 3);
  int colD = nt * 16 + (lane & 15);
#pragma unroll
  for (int d = 0; d < 8; ++d)
    out[(size_t)(rowD + d) * N + colD] = c[d];
}

// LSTM nonlinearity over raw gates (B x 4H), torch order i,f,g,o
__global__ void lstm_pointwise(const float* __restrict__ gates,
                               const float* __restrict__ bias,
                               float* __restrict__ cst,
                               float* __restrict__ hout, int H) {
  int idx = blockIdx.x * blockDim.x + threadIdx.x;
  if (idx >= B * H) return;
  int b = idx / H, h = idx - b * H;
  const float* g = gates + (size_t)b * 4 * H;
  float i  = g[h]         + bias[h];
  float f  = g[H + h]     + bias[H + h];
  float gg = g[2 * H + h] + bias[2 * H + h];
  float o  = g[3 * H + h] + bias[3 * H + h];
  float c = sigf(f) * cst[idx] + sigf(i) * tanhf(gg);
  cst[idx] = c;
  hout[idx] = sigf(o) * tanhf(c);
}

// q = ah @ query_w.T  (B x ATT_H)
__global__ void query_kernel(const float* __restrict__ ah,
                             const float* __restrict__ qw,
                             float* __restrict__ q) {
  int b = blockIdx.x, h = threadIdx.x;   // grid B, block ATT_H
  const float* a = ah + (size_t)b * HID;
  const float* w = qw + (size_t)h * HID;
  float s = 0.f;
  for (int k = 0; k < HID; ++k) s += a[k] * w[k];
  q[b * ATT_H + h] = s;
}

// location conv + loc dense + energies. grid = B*4 (64 positions each), block 256.
__global__ void attn_energies(const float* __restrict__ aw,
                              const float* __restrict__ awc,
                              const float* __restrict__ convw,   // (32,2,31)
                              const float* __restrict__ ldw,     // (128,32)
                              const float* __restrict__ q,
                              const float* __restrict__ pmem,
                              const float* __restrict__ vw,      // (128)
                              float* __restrict__ energ) {
  int b = blockIdx.x >> 2, pg = blockIdx.x & 3;
  __shared__ float s_aw[T_ENC], s_awc[T_ENC], s_conv[64][N_FILT], s_part[256];
  int tid = threadIdx.x;
  s_aw[tid]  = aw[b * T_ENC + tid];
  s_awc[tid] = awc[b * T_ENC + tid];
  __syncthreads();
  int t = pg * 64 + (tid >> 2);
  int sub = tid & 3;
  // each thread: 8 filters of the conv at position t
  for (int f = sub * 8; f < sub * 8 + 8; ++f) {
    float s = 0.f;
    for (int k = 0; k < KSZ; ++k) {
      int p = t + k - (KSZ / 2);
      if (p >= 0 && p < T_ENC) {
        s += convw[(f * 2 + 0) * KSZ + k] * s_aw[p];
        s += convw[(f * 2 + 1) * KSZ + k] * s_awc[p];
      }
    }
    s_conv[t - pg * 64][f] = s;
  }
  __syncthreads();
  // each thread: 32 of the 128 hidden dims -> partial energy
  float e = 0.f;
  const float* pm = pmem + (size_t)(b * T_ENC + t) * ATT_H;
  const float* qb = q + b * ATT_H;
  for (int h = sub * 32; h < sub * 32 + 32; ++h) {
    float d = 0.f;
    for (int f = 0; f < N_FILT; ++f) d += s_conv[t - pg * 64][f] * ldw[h * N_FILT + f];
    e += vw[h] * tanhf(qb[h] + d + pm[h]);
  }
  s_part[tid] = e;
  __syncthreads();
  if (sub == 0)
    energ[b * T_ENC + t] = s_part[tid] + s_part[tid + 1] + s_part[tid + 2] + s_part[tid + 3];
}

// masked softmax over T_ENC, context = aw @ memory, awcum update, alignment out
__global__ void attn_softmax_ctx(const float* __restrict__ energ,
                                 const int* __restrict__ lens,
                                 const float* __restrict__ memory,
                                 float* __restrict__ aw,
                                 float* __restrict__ awc,
                                 float* __restrict__ actx,
                                 float* __restrict__ align_t) {  // base + t*T_ENC
  int b = blockIdx.x, tid = threadIdx.x;  // block 256
  __shared__ float s_w[T_ENC], s_red[256];
  int len = lens[b];
  float e = (tid < len) ? energ[b * T_ENC + tid] : -1e30f;
  s_red[tid] = e;
  __syncthreads();
  for (int s = 128; s > 0; s >>= 1) {
    if (tid < s) s_red[tid] = fmaxf(s_red[tid], s_red[tid + s]);
    __syncthreads();
  }
  float mx = s_red[0];
  __syncthreads();
  float ex = (tid < len) ? expf(e - mx) : 0.f;
  s_red[tid] = ex;
  __syncthreads();
  for (int s = 128; s > 0; s >>= 1) {
    if (tid < s) s_red[tid] += s_red[tid + s];
    __syncthreads();
  }
  float w = ex / s_red[0];
  s_w[tid] = w;
  aw[b * T_ENC + tid] = w;
  awc[b * T_ENC + tid] += w;
  align_t[(size_t)b * T_OUT * T_ENC + tid] = w;
  __syncthreads();
  for (int e2 = tid; e2 < E_DIM; e2 += 256) {
    float s = 0.f;
    const float* mb = memory + (size_t)b * T_ENC * E_DIM + e2;
    for (int t2 = 0; t2 < T_ENC; ++t2) s += s_w[t2] * mb[(size_t)t2 * E_DIM];
    actx[b * E_DIM + e2] = s;
  }
}

// mel + gate projection from [dh | ctx]
__global__ void proj_kernel(const float* __restrict__ dh,
                            const float* __restrict__ actx,
                            const float* __restrict__ projw, const float* __restrict__ projb,
                            const float* __restrict__ gatew, const float* __restrict__ gateb,
                            float* __restrict__ mel_out, float* __restrict__ gate_out, int t) {
  int b = blockIdx.x, j = threadIdx.x;   // block 128; 0..79 mel, 80 gate
  if (j > N_MELS) return;
  const float* w = (j < N_MELS) ? (projw + (size_t)j * (HID + E_DIM)) : gatew;
  float s = (j < N_MELS) ? projb[j] : gateb[0];
  const float* d = dh + (size_t)b * HID;
  const float* a = actx + (size_t)b * E_DIM;
  for (int k = 0; k < HID; ++k) s += d[k] * w[k];
  for (int k = 0; k < E_DIM; ++k) s += a[k] * w[HID + k];
  if (j < N_MELS) mel_out[(size_t)b * N_MELS * T_OUT + j * T_OUT + t] = s;
  else            gate_out[b * T_OUT + t] = s;
}

// ---------------- host ----------------
extern "C" void kernel_launch(void* const* d_in, const int* in_sizes, int n_in,
                              void* d_out, int out_size, void* d_ws, size_t ws_size,
                              hipStream_t stream) {
  const float* memory   = (const float*)d_in[0];
  const float* mel      = (const float*)d_in[1];
  const int*   lens     = (const int*)d_in[2];
  const float* pw1      = (const float*)d_in[3];
  const float* pw2      = (const float*)d_in[4];
  const float* attn_wih = (const float*)d_in[5];
  const float* attn_whh = (const float*)d_in[6];
  const float* attn_b   = (const float*)d_in[7];
  const float* query_w  = (const float*)d_in[8];
  const float* memory_w = (const float*)d_in[9];
  const float* v_w      = (const float*)d_in[10];
  const float* conv_w   = (const float*)d_in[11];
  const float* ldense_w = (const float*)d_in[12];
  const float* dec_wih  = (const float*)d_in[13];
  const float* dec_whh  = (const float*)d_in[14];
  const float* dec_b    = (const float*)d_in[15];
  const float* proj_w   = (const float*)d_in[16];
  const float* proj_b   = (const float*)d_in[17];
  const float* gate_w   = (const float*)d_in[18];
  const float* gate_b   = (const float*)d_in[19];
  (void)in_sizes; (void)n_in; (void)out_size;

  char* base = (char*)d_ws;
  size_t off = 0;
  auto alloc = [&](size_t bytes) -> void* {
    off = (off + 255) & ~(size_t)255;
    void* p = base + off;
    off += bytes;
    return p;
  };
  const int STATE_FLOATS = 4 * B * HID + 2 * B * T_ENC + B * E_DIM;  // 163840
  unsigned short* wattn = (unsigned short*)alloc((size_t)NG * KATTN * 2);
  unsigned short* wdec  = (unsigned short*)alloc((size_t)NG * KDEC * 2);
  unsigned short* pren  = (unsigned short*)alloc((size_t)T_OUT * B * PRE * 2);
  float* pmem  = (float*)alloc((size_t)B * T_ENC * ATT_H * 4);
  float* gates = (float*)alloc((size_t)B * NG * 4);
  unsigned short* xa = (unsigned short*)alloc((size_t)B * KATTN * 2);
  unsigned short* xd = (unsigned short*)alloc((size_t)B * KDEC * 2);
  float* states = (float*)alloc((size_t)STATE_FLOATS * 4);
  float* q     = (float*)alloc((size_t)B * ATT_H * 4);
  float* energ = (float*)alloc((size_t)B * T_ENC * 4);
  if (off > ws_size) return;  // workspace too small; nothing safe to do

  float* ah  = states;
  float* ac  = ah + B * HID;
  float* dh  = ac + B * HID;
  float* dc  = dh + B * HID;
  float* aw  = dc + B * HID;
  float* awc = aw + B * T_ENC;
  float* actx = awc + B * T_ENC;

  float* mel_out   = (float*)d_out;
  float* gate_out  = mel_out + (size_t)B * N_MELS * T_OUT;
  float* align_out = gate_out + (size_t)B * T_OUT;

  // one-time prep (re-run every call; deterministic, graph-capturable)
  hipLaunchKernelGGL(concat_bf16, dim3(2048), dim3(256), 0, stream,
                     attn_wih, PRE + E_DIM, attn_whh, HID, wattn, NG);
  hipLaunchKernelGGL(concat_bf16, dim3(2048), dim3(256), 0, stream,
                     dec_wih, HID + E_DIM, dec_whh, HID, wdec, NG);
  hipLaunchKernelGGL(prenet_kernel, dim3(T_OUT * B), dim3(256), 0, stream, mel, pw1, pw2, pren);
  hipLaunchKernelGGL(pmem_kernel, dim3(B * T_ENC), dim3(ATT_H), 0, stream, memory, memory_w, pmem);
  hipLaunchKernelGGL(zero_kernel, dim3(256), dim3(256), 0, stream, states, STATE_FLOATS);

  const int gemmBlocks = (2 * (NG / 16) * 32) / 256;  // 64 blocks, 512 waves
  for (int t = 0; t < T_OUT; ++t) {
    hipLaunchKernelGGL(assemble_xa, dim3((B * KATTN + 255) / 256), dim3(256), 0, stream,
                       pren + (size_t)t * B * PRE, actx, ah, xa);
    hipLaunchKernelGGL(gemm32_bf16_wmma, dim3(gemmBlocks), dim3(256), 0, stream,
                       xa, wattn, gates, KATTN, NG);
    hipLaunchKernelGGL(lstm_pointwise, dim3((B * HID + 255) / 256), dim3(256), 0, stream,
                       gates, attn_b, ac, ah, HID);
    hipLaunchKernelGGL(query_kernel, dim3(B), dim3(ATT_H), 0, stream, ah, query_w, q);
    hipLaunchKernelGGL(attn_energies, dim3(B * 4), dim3(256), 0, stream,
                       aw, awc, conv_w, ldense_w, q, pmem, v_w, energ);
    hipLaunchKernelGGL(attn_softmax_ctx, dim3(B), dim3(256), 0, stream,
                       energ, lens, memory, aw, awc, actx, align_out + (size_t)t * T_ENC);
    hipLaunchKernelGGL(assemble_xd, dim3((B * KDEC + 255) / 256), dim3(256), 0, stream,
                       ah, actx, dh, xd);
    hipLaunchKernelGGL(gemm32_bf16_wmma, dim3(gemmBlocks), dim3(256), 0, stream,
                       xd, wdec, gates, KDEC, NG);
    hipLaunchKernelGGL(lstm_pointwise, dim3((B * HID + 255) / 256), dim3(256), 0, stream,
                       gates, dec_b, dc, dh, HID);
    hipLaunchKernelGGL(proj_kernel, dim3(B), dim3(128), 0, stream,
                       dh, actx, proj_w, proj_b, gate_w, gate_b, mel_out, gate_out, t);
  }
}
